// RNN_Decoder_72138270703952
// MI455X (gfx1250) — compile-verified, matrix operand order checked by reading, not run on hardware
//
#include <hip/hip_runtime.h>

typedef __attribute__((ext_vector_type(16))) _Float16 v16h;
typedef __attribute__((ext_vector_type(8)))  float    v8f;
typedef __attribute__((ext_vector_type(4)))  unsigned int v4u;

union H16 { v16h h; v4u u[2]; };

#define NROWS   12800
#define HID     256
#define GATES   1024
#define T_STEPS 16
#define MROWS   32          // rows per workgroup
#define ROWH    264         // padded LDS row length in halfs (528 B, 16B-aligned, bank-friendly)
#define KTILES  8           // 256 / 32

// ---------------- weight packing: f32 -> f16 in WMMA B-operand lane layout ----------------
// Packed layout: [ntile(64)][ktile(8)][lane(32)][16 halfs]  -> lane loads contiguous 32 B.
// B (K x N) layout mirror of the documented 16-bit A layout:
//   lanes 0-15: N = lane,    K = kt*32 + (j<8 ? j : j+8)
//   lanes 16-31: N = lane-16, K = kt*32 + 8 + (j<8 ? j : j+8)
// B[k][n] = W[n*HID + k]  (since gates = hx @ W^T).
__global__ void pack_weights(const float* __restrict__ Wih, const float* __restrict__ Whh,
                             _Float16* __restrict__ pIh, _Float16* __restrict__ pHh) {
    int t    = blockIdx.x * blockDim.x + threadIdx.x;   // 0 .. 32767
    int lane = t & 31;
    int kt   = (t >> 5) & 7;
    int nt   = (t >> 8) & 63;
    int w    = t >> 14;                                  // 0 = W_ih, 1 = W_hh
    const float* W = w ? Whh : Wih;
    _Float16*    P = w ? pHh : pIh;
    int n     = nt * 16 + (lane & 15);
    int kbase = kt * 32 + ((lane >> 4) << 3);
    H16 o;
    #pragma unroll
    for (int j = 0; j < 16; ++j) {
        int k = kbase + (j < 8 ? j : j + 8);
        o.h[j] = (_Float16)W[n * HID + k];
    }
    v4u* dst = (v4u*)(P + (size_t)((nt * KTILES + kt) * 32 + lane) * 16);
    dst[0] = o.u[0];
    dst[1] = o.u[1];
}

__global__ void pack_bias(const float* __restrict__ bi, const float* __restrict__ bh,
                          float* __restrict__ bsum) {
    int i = blockIdx.x * blockDim.x + threadIdx.x;
    if (i < GATES) bsum[i] = bi[i] + bh[i];
}

// ---------------- device helpers ----------------
__device__ __forceinline__ float sigf(float x) { return 1.0f / (1.0f + __expf(-x)); }
__device__ __forceinline__ float tanhfast(float x) {
    float ax = fabsf(x);
    float e  = __expf(-2.0f * ax);
    float r  = (1.0f - e) / (1.0f + e);
    return copysignf(r, x);
}

// One full [32 x 256] x [256 x 1024] GEMM contribution for this wave's 8 N-tiles, 2 M-tiles.
// A (f16) from LDS, B (f16) from pre-packed global, accumulate f32.
__device__ __forceinline__ void gemm_step(const _Float16* __restrict__ pW,
                                          const _Float16* abuf,
                                          int wave, int lane, v8f acc[2][8]) {
    const int r  = lane & 15;
    const int ko = (lane >> 4) << 3;   // 0 or 8 half offset per half-wave
    #pragma unroll
    for (int kt = 0; kt < KTILES; ++kt) {
        int kb = kt * 32 + ko;
        H16 a0, a1;
        a0.u[0] = *(const v4u*)&abuf[r * ROWH + kb];
        a0.u[1] = *(const v4u*)&abuf[r * ROWH + kb + 16];
        a1.u[0] = *(const v4u*)&abuf[(16 + r) * ROWH + kb];
        a1.u[1] = *(const v4u*)&abuf[(16 + r) * ROWH + kb + 16];
        #pragma unroll
        for (int nt = 0; nt < 8; ++nt) {
            int g = nt >> 1, c = nt & 1;
            int ntg = g * 16 + wave * 2 + c;              // global 16-col tile index in [0,64)
            const v4u* pb = (const v4u*)(pW + (size_t)((ntg * KTILES + kt) * 32 + lane) * 16);
            H16 b;
            b.u[0] = pb[0];
            b.u[1] = pb[1];
            acc[0][nt] = __builtin_amdgcn_wmma_f32_16x16x32_f16(
                false, a0.h, false, b.h, (short)0, acc[0][nt], false, false);
            acc[1][nt] = __builtin_amdgcn_wmma_f32_16x16x32_f16(
                false, a1.h, false, b.h, (short)0, acc[1][nt], false, false);
        }
    }
}

// ---------------- fused LSTM decoder kernel ----------------
__global__ __launch_bounds__(256, 2) void lstm_fused(
    const float* __restrict__ x,      // (8,256,40,40)
    const float* __restrict__ hx0,    // (12800,256)
    const float* __restrict__ cx0,    // (12800,256)
    const _Float16* __restrict__ pIh, // packed W_ih f16
    const _Float16* __restrict__ pHh, // packed W_hh f16
    const float* __restrict__ bias,   // b_ih + b_hh (1024)
    const float* __restrict__ Wlin,   // (2,256)
    const float* __restrict__ blin,   // (2)
    float* __restrict__ out)          // (8,40,40,16,2) flat = n*32 + t*2 + o
{
    __shared__ __align__(16) _Float16 abuf[MROWS * ROWH];   // hx / xf tile (f16)
    __shared__ float wl[2 * HID];                            // W_lin
    __shared__ float ypart[MROWS * 8 * 2];                   // head partial sums

    const int tid  = threadIdx.x;
    const int wave = tid >> 5;
    const int lane = tid & 31;
    const int R0   = blockIdx.x * MROWS;

    // W_lin -> LDS
    wl[tid]       = Wlin[tid];
    wl[256 + tid] = Wlin[256 + tid];

    // ---- Phase A: gather xf tile (permute(0,2,3,1)) into LDS as f16, coalesced along hw ----
    {
        int bs  = R0 / 1600;
        int hw0 = R0 - bs * 1600;          // MROWS=32 divides 1600 -> whole tile in one bs
        int row = tid & 31;
        int cw  = tid >> 5;
        for (int cc = 0; cc < 32; ++cc) {
            int c = cc * 8 + cw;
            float v = x[(size_t)(bs * HID + c) * 1600 + hw0 + row];
            abuf[row * ROWH + c] = (_Float16)v;
        }
    }
    __syncthreads();

    // ---- Phase B: x_proj = xf @ W_ih^T + (b_ih + b_hh), held in registers ----
    v8f xacc[2][8];
    #pragma unroll
    for (int mt = 0; mt < 2; ++mt)
        #pragma unroll
        for (int nt = 0; nt < 8; ++nt)
            #pragma unroll
            for (int r = 0; r < 8; ++r) xacc[mt][nt][r] = 0.0f;

    gemm_step(pIh, abuf, wave, lane, xacc);

    #pragma unroll
    for (int nt = 0; nt < 8; ++nt) {
        int g = nt >> 1, c = nt & 1;
        int col = g * 256 + wave * 32 + c * 16 + (lane & 15);
        float b = bias[col];
        #pragma unroll
        for (int r = 0; r < 8; ++r) { xacc[0][nt][r] += b; xacc[1][nt][r] += b; }
    }
    __syncthreads();   // done reading xf from abuf

    // ---- Phase C: hx -> LDS (f16), cx -> registers in C-matrix layout ----
    for (int i = 0; i < 32; ++i) {
        int idx = i * 256 + tid;
        int row = idx >> 8;
        int c   = idx & 255;
        abuf[row * ROWH + c] = (_Float16)hx0[(size_t)(R0 + row) * HID + c];
    }
    v8f cxr[2][2];
    #pragma unroll
    for (int mt = 0; mt < 2; ++mt)
        #pragma unroll
        for (int c = 0; c < 2; ++c)
            #pragma unroll
            for (int r = 0; r < 8; ++r) {
                int row = mt * 16 + r + ((lane >> 4) << 3);
                int col = wave * 32 + c * 16 + (lane & 15);
                cxr[mt][c][r] = cx0[(size_t)(R0 + row) * HID + col];
            }
    __syncthreads();

    // ---- Phase D: 16 recurrent steps, fully fused ----
    for (int t = 0; t < T_STEPS; ++t) {
        v8f acc[2][8];
        #pragma unroll
        for (int mt = 0; mt < 2; ++mt)
            #pragma unroll
            for (int nt = 0; nt < 8; ++nt) acc[mt][nt] = xacc[mt][nt];

        gemm_step(pHh, abuf, wave, lane, acc);
        __syncthreads();   // all waves done reading old h before overwrite

        // LSTM elementwise: wave holds i/f/g/o for its own hidden columns in identical layout
        #pragma unroll
        for (int mt = 0; mt < 2; ++mt)
            #pragma unroll
            for (int c = 0; c < 2; ++c)
                #pragma unroll
                for (int r = 0; r < 8; ++r) {
                    float ig = sigf(acc[mt][0 + c][r]);
                    float fg = sigf(acc[mt][2 + c][r]);
                    float gg = tanhfast(acc[mt][4 + c][r]);
                    float og = sigf(acc[mt][6 + c][r]);
                    float cn = fg * cxr[mt][c][r] + ig * gg;
                    cxr[mt][c][r] = cn;
                    float h = og * tanhfast(cn);
                    int row = mt * 16 + r + ((lane >> 4) << 3);
                    int col = wave * 32 + c * 16 + (lane & 15);
                    abuf[row * ROWH + col] = (_Float16)h;
                }
        __syncthreads();   // new h visible to all waves

        // Output head: y[t,n,:] = leaky_relu(h) @ W_lin^T + b_lin (O = 2)
        {
            int lr  = tid >> 3;
            int sub = tid & 7;
            float s0 = 0.0f, s1 = 0.0f;
            #pragma unroll 8
            for (int jj = 0; jj < 32; ++jj) {
                int j   = sub * 32 + jj;
                float hv = (float)abuf[lr * ROWH + j];
                hv = (hv > 0.0f) ? hv : 0.01f * hv;
                s0 += hv * wl[j];
                s1 += hv * wl[256 + j];
            }
            ypart[(lr * 8 + sub) * 2 + 0] = s0;
            ypart[(lr * 8 + sub) * 2 + 1] = s1;
        }
        __syncthreads();
        if (tid < 64) {
            int lr = tid >> 1, o = tid & 1;
            float s = blin[o];
            #pragma unroll
            for (int sub = 0; sub < 8; ++sub) s += ypart[(lr * 8 + sub) * 2 + o];
            out[(size_t)(R0 + lr) * 32 + t * 2 + o] = s;
        }
    }
}

// ---------------- host launcher ----------------
extern "C" void kernel_launch(void* const* d_in, const int* in_sizes, int n_in,
                              void* d_out, int out_size, void* d_ws, size_t ws_size,
                              hipStream_t stream) {
    const float* x    = (const float*)d_in[0];
    const float* hx   = (const float*)d_in[1];
    const float* cx   = (const float*)d_in[2];
    const float* Wih  = (const float*)d_in[3];
    const float* Whh  = (const float*)d_in[4];
    const float* bih  = (const float*)d_in[5];
    const float* bhh  = (const float*)d_in[6];
    const float* Wlin = (const float*)d_in[7];
    const float* blin = (const float*)d_in[8];
    float* out = (float*)d_out;

    char* ws = (char*)d_ws;
    _Float16* pHh  = (_Float16*)(ws);                 // 512 KB
    _Float16* pIh  = (_Float16*)(ws + 512 * 1024);    // 512 KB
    float*    bsum = (float*)(ws + 1024 * 1024);      // 4 KB

    pack_weights<<<32768 / 256, 256, 0, stream>>>(Wih, Whh, pIh, pHh);
    pack_bias<<<4, 256, 0, stream>>>(bih, bhh, bsum);
    lstm_fused<<<NROWS / MROWS, 256, 0, stream>>>(x, hx, cx, pIh, pHh, bsum, Wlin, blin, out);
}
